// DisGraphRep_65068754534603
// MI455X (gfx1250) — matrix operand order repaired and checked by reading, not compiled, not run on hardware
//
#include <hip/hip_runtime.h>

typedef __attribute__((ext_vector_type(2)))  float    v2f;
typedef __attribute__((ext_vector_type(8)))  float    v8f;
typedef __attribute__((ext_vector_type(16))) _Float16 v16h;

#define HD 128   // hid_dim
#define DDIM 64  // dist_embed_dim

// ---------------- preamble kernels ----------------

__global__ __launch_bounds__(256) void k_init(const float* __restrict__ poi,
                                              float* __restrict__ x, float* __restrict__ acc,
                                              float* __restrict__ deg, int N, long long ND) {
  long long i = (long long)blockIdx.x * blockDim.x + threadIdx.x;
  if (i < ND) { float v = poi[i]; x[i] = v; acc[i] = v; }
  if (i < N) deg[i] = 1.0f;  // self-loop contributes 1 to every node's in-degree
}

__global__ __launch_bounds__(256) void k_deg(const long long* __restrict__ ei,
                                             float* __restrict__ deg, int E) {
  int e = blockIdx.x * blockDim.x + threadIdx.x;
  if (e < E) {
    int d = (int)ei[(long long)E + e];  // dst row of edge_index
    unsafeAtomicAdd(&deg[d], 1.0f);
  }
}

__global__ __launch_bounds__(256) void k_dinv(const float* __restrict__ deg,
                                              float* __restrict__ dinv, int N) {
  int v = blockIdx.x * blockDim.x + threadIdx.x;
  if (v < N) { float d = deg[v]; dinv[v] = d > 0.0f ? rsqrtf(d) : 0.0f; }
}

// ---------------- node GEMM: y = x @ W^T + b  (f32 WMMA 16x16x4) ----------------
// The 16x128 x-tile (8KB, contiguous) is staged into LDS once per block with
// GLOBAL_LOAD_ASYNC_TO_LDS_B128 (ASYNCcnt), then all 8 waves read A from LDS.
// One wave computes a 16x16 output tile; 8 waves/block cover all 128 columns.

__global__ __launch_bounds__(256) void gemm_xW(const float* __restrict__ x,
                                               const float* __restrict__ W,
                                               const float* __restrict__ b,
                                               float* __restrict__ y, int nrows) {
  __shared__ float xs[16 * HD];
  int tid  = threadIdx.x;
  int wave = tid >> 5;
  int lane = tid & 31;
  int n = lane & 15, hi = lane >> 4;
  int rowbase = blockIdx.x * 16;
  if (rowbase >= nrows) return;

  // ---- async DMA stage: 2 x B128 chunks per thread = 8KB tile ----
  long long tilebase = (long long)rowbase * HD;         // float index of tile start
  long long maxf = (long long)nrows * HD - 4;           // last valid float4 start
#pragma unroll
  for (int c = 0; c < 2; ++c) {
    int chunk = tid + c * 256;                          // 0..511 sixteen-byte chunks
    long long fofs = tilebase + (long long)chunk * 4;
    if (fofs > maxf) fofs = maxf;                       // clamp (inert for exact tiles)
    const float* gp = x + fofs;
    unsigned lds = (unsigned)(size_t)(&xs[chunk * 4]);  // workgroup-relative LDS offset
    asm volatile("global_load_async_to_lds_b128 %0, %1, off"
                 :: "v"(lds), "v"(gp) : "memory");
  }
  asm volatile("s_wait_asynccnt 0x0" ::: "memory");
  __syncthreads();

  int col = wave * 16 + n;
  float bias = b[col];
  v8f acc = {bias, bias, bias, bias, bias, bias, bias, bias};
  const float* xr = &xs[n * HD];              // A-matrix row (LDS), same in both lane halves
  const float* wr = W + (long long)col * HD;  // row `col` of W; B[k,n] = W[col,k]
  for (int ks = 0; ks < HD / 4; ++ks) {
    int k = ks * 4 + 2 * hi;
    v2f a  = *(const v2f*)(xr + k);   // ds_load_b64: A (M=n, K=k..k+1)
    v2f bb = *(const v2f*)(wr + k);   // B: (K=k, N=n), (K=k+1, N=n)
    acc = __builtin_amdgcn_wmma_f32_16x16x4_f32(false, a, false, bb, (short)0, acc, false, false);
  }
  // D element r -> y[rowbase + r + 8*hi, col]
#pragma unroll
  for (int r = 0; r < 8; ++r) {
    int row = rowbase + r + 8 * hi;
    if (row < nrows) y[(long long)row * HD + col] = acc[r];
  }
}

// ---------------- self-loop dist vector: dwself[d] = d2b[d] + sum_k relu(d1W[k]+d1b[k])*d2W[d,k] ----

__global__ void k_dwself(const float* __restrict__ d1W, const float* __restrict__ d1b,
                         const float* __restrict__ d2W, const float* __restrict__ d2b,
                         float* __restrict__ dwself) {
  int d = threadIdx.x;
  if (d < HD) {
    float s = d2b[d];
    const float* wr = d2W + d * DDIM;
    for (int k = 0; k < DDIM; ++k) {
      float h1 = d1W[k] + d1b[k];  // ew = exp(0) = 1 for self loops
      h1 = h1 > 0.0f ? h1 : 0.0f;
      s += h1 * wr[k];
    }
    dwself[d] = s;
  }
}

// h initialized with the self-loop message (saves a zero pass)
__global__ __launch_bounds__(256) void k_inith(const float* __restrict__ y,
                                               const float* __restrict__ dinv,
                                               const float* __restrict__ dwself,
                                               float* __restrict__ h, long long ND) {
  long long i = (long long)blockIdx.x * blockDim.x + threadIdx.x;
  if (i < ND) {
    int v = (int)(i >> 7), d = (int)(i & (HD - 1));
    float di = dinv[v];
    h[i] = di * di * y[i] * dwself[d];
  }
}

// ---------------- fused edge kernel: dist-MLP via f16 WMMA + gather + scatter-add --------------
// Each wave owns one 16-column slab (nt) of the output and streams 16-edge tiles.
// A (16 edges x 64) = relu(ew * d1W + d1b) built in registers (packed f16),
// B (64 x 16) = d2W^T slab, resident in v16h registers, split into two K=32 WMMAs.

__global__ __launch_bounds__(256) void edge_msg(const long long* __restrict__ ei,
                                                const float* __restrict__ dist,
                                                const float* __restrict__ dinv,
                                                const float* __restrict__ y,
                                                float* __restrict__ h,
                                                const float* __restrict__ d1W,
                                                const float* __restrict__ d1b,
                                                const float* __restrict__ d2W,
                                                const float* __restrict__ d2b,
                                                int E, int ntiles) {
  int lane = threadIdx.x & 31;
  int wgl  = blockIdx.x * 8 + (threadIdx.x >> 5);
  int nt   = wgl & 7;                 // column tile 0..7
  int colbase = nt * 16;
  int n = lane & 15, hi = lane >> 4;

  // A-side constants in 16-bit A 16x32 layout: VGPR p<4 -> K=2p+8hi, p>=4 -> K=16+8hi+2(p-4)
  v16h av0, bv0, av1, bv1;
#pragma unroll
  for (int p = 0; p < 8; ++p) {
    int K = (p < 4) ? (2 * p + 8 * hi) : (16 + 8 * hi + 2 * (p - 4));
    av0[2 * p]     = (_Float16)d1W[K];
    av0[2 * p + 1] = (_Float16)d1W[K + 1];
    bv0[2 * p]     = (_Float16)d1b[K];
    bv0[2 * p + 1] = (_Float16)d1b[K + 1];
    av1[2 * p]     = (_Float16)d1W[K + 32];
    av1[2 * p + 1] = (_Float16)d1W[K + 33];
    bv1[2 * p]     = (_Float16)d1b[K + 32];
    bv1[2 * p + 1] = (_Float16)d1b[K + 33];
  }
  // B tiles: B[k, n] = d2W[colbase+n, k]; 16-bit B 32x16: element i<8 -> K=8hi+i, i>=8 -> K=16+8hi+(i-8)
  v16h B0, B1;
  const float* wrow = d2W + (long long)(colbase + n) * DDIM;
#pragma unroll
  for (int i = 0; i < 16; ++i) {
    int kb = (i < 8) ? (i + 8 * hi) : (8 + i + 8 * hi);
    B0[i] = (_Float16)wrow[kb];
    B1[i] = (_Float16)wrow[kb + 32];
  }
  float cbias = d2b[colbase + n];

  int stride = gridDim.x;  // (gridDim.x * 8 waves) / 8 slabs
  for (int t = wgl >> 3; t < ntiles; t += stride) {
    int m = lane & 15;
    int e = t * 16 + m;
    int ec = e < E ? e : 0;
    int se = (int)ei[ec];
    int de = (int)ei[(long long)E + ec];
    float dv = dist[ec];
    float ew = __expf(-dv * dv);
    float nrm = (e < E) ? dinv[se] * dinv[de] : 0.0f;

    _Float16 ewh = (_Float16)ew;
    v16h A0, A1;
#pragma unroll
    for (int i = 0; i < 16; ++i) {
      _Float16 t0 = ewh * av0[i] + bv0[i];
      _Float16 t1 = ewh * av1[i] + bv1[i];
      A0[i] = t0 > (_Float16)0 ? t0 : (_Float16)0;
      A1[i] = t1 > (_Float16)0 ? t1 : (_Float16)0;
    }
    v8f acc = {cbias, cbias, cbias, cbias, cbias, cbias, cbias, cbias};
    acc = __builtin_amdgcn_wmma_f32_16x16x32_f16(false, A0, false, B0, (short)0, acc, false, false);
    acc = __builtin_amdgcn_wmma_f32_16x16x32_f16(false, A1, false, B1, (short)0, acc, false, false);

    // D element r belongs to edge (r + 8*hi); fetch that edge's scalars cross-lane (ds_bpermute)
#pragma unroll
    for (int r = 0; r < 8; ++r) {
      int   ml = r + 8 * hi;
      int   s2 = __shfl(se, ml, 32);
      int   d2 = __shfl(de, ml, 32);
      float nm = __shfl(nrm, ml, 32);
      float yv = y[(long long)s2 * HD + colbase + n];
      float msg = nm * yv * acc[r];
      unsafeAtomicAdd(&h[(long long)d2 * HD + colbase + n], msg);
    }
  }
}

// ---------------- activation + accumulate, final scale ----------------

__global__ __launch_bounds__(256) void k_act(const float* __restrict__ h,
                                             float* __restrict__ x, float* __restrict__ acc,
                                             long long ND) {
  long long i = (long long)blockIdx.x * blockDim.x + threadIdx.x;
  if (i < ND) {
    float v = h[i];
    float o = v > 0.0f ? v : 0.01f * v;
    x[i] = o;
    acc[i] += o;
  }
}

__global__ __launch_bounds__(256) void k_final(const float* __restrict__ acc,
                                               float* __restrict__ out, long long ND, float s) {
  long long i = (long long)blockIdx.x * blockDim.x + threadIdx.x;
  if (i < ND) out[i] = acc[i] * s;
}

// ---------------- driver ----------------

extern "C" void kernel_launch(void* const* d_in, const int* in_sizes, int n_in,
                              void* d_out, int out_size, void* d_ws, size_t ws_size,
                              hipStream_t stream) {
  const float*     poi  = (const float*)d_in[0];
  const long long* ei   = (const long long*)d_in[1];
  const float*     dist = (const float*)d_in[2];
  const float*     linW = (const float*)d_in[3];
  const float*     linb = (const float*)d_in[4];
  const float*     d1W  = (const float*)d_in[5];
  const float*     d1b  = (const float*)d_in[6];
  const float*     d2W  = (const float*)d_in[7];
  const float*     d2b  = (const float*)d_in[8];

  const int L = 2;
  const int N = in_sizes[0] / HD;
  const int E = in_sizes[2];
  const long long ND = (long long)N * HD;

  float* ws     = (float*)d_ws;
  float* x      = ws;
  float* acc    = ws + ND;
  float* y      = ws + 2 * ND;
  float* h      = ws + 3 * ND;
  float* deg    = ws + 4 * ND;
  float* dinv   = deg + N;
  float* dwself = dinv + N;

  int bND = (int)((ND + 255) / 256);
  int ntiles = (E + 15) / 16;

  k_init<<<bND, 256, 0, stream>>>(poi, x, acc, deg, N, ND);
  k_deg<<<(E + 255) / 256, 256, 0, stream>>>(ei, deg, E);
  k_dinv<<<(N + 255) / 256, 256, 0, stream>>>(deg, dinv, N);

  for (int l = 0; l < L; ++l) {
    gemm_xW<<<(N + 15) / 16, 256, 0, stream>>>(x, linW + (long long)l * HD * HD,
                                               linb + (long long)l * HD, y, N);
    k_dwself<<<1, 128, 0, stream>>>(d1W + l * DDIM, d1b + l * DDIM,
                                    d2W + (long long)l * HD * DDIM, d2b + (long long)l * HD,
                                    dwself);
    k_inith<<<bND, 256, 0, stream>>>(y, dinv, dwself, h, ND);
    edge_msg<<<2048, 256, 0, stream>>>(ei, dist, dinv, y, h,
                                       d1W + l * DDIM, d1b + l * DDIM,
                                       d2W + (long long)l * HD * DDIM, d2b + (long long)l * HD,
                                       E, ntiles);
    k_act<<<bND, 256, 0, stream>>>(h, x, acc, ND);
  }
  k_final<<<bND, 256, 0, stream>>>(acc, (float*)d_out, ND, 1.0f / (L + 1));
}
